// Net_20452634264014
// MI455X (gfx1250) — matrix operand order repaired and checked by reading, not compiled, not run on hardware
//
#include <hip/hip_runtime.h>
#include <hip/hip_bf16.h>
#include <math.h>

typedef __attribute__((ext_vector_type(16))) _Float16 v16h;
typedef __attribute__((ext_vector_type(8)))  float    v8f;

#define B_   8
#define N_   16384
#define M1_  328
#define M2_  82
#define NB_  64
#define NC_  13
#define NCP_ 16     // padded class count

static inline unsigned gridFor(size_t n, int b) { return (unsigned)((n + (size_t)b - 1) / (size_t)b); }

__device__ __forceinline__ v16h cvt_a16(float4 a0, float4 a1, float4 a2, float4 a3)
{
    v16h af;
    af[0]=(_Float16)a0.x;  af[1]=(_Float16)a0.y;  af[2]=(_Float16)a0.z;  af[3]=(_Float16)a0.w;
    af[4]=(_Float16)a1.x;  af[5]=(_Float16)a1.y;  af[6]=(_Float16)a1.z;  af[7]=(_Float16)a1.w;
    af[8]=(_Float16)a2.x;  af[9]=(_Float16)a2.y;  af[10]=(_Float16)a2.z; af[11]=(_Float16)a2.w;
    af[12]=(_Float16)a3.x; af[13]=(_Float16)a3.y; af[14]=(_Float16)a3.z; af[15]=(_Float16)a3.w;
    return af;
}

// ---------------------------------------------------------------------------
// Farthest point sampling: one block per batch cloud.
// ---------------------------------------------------------------------------
__global__ void fps_kernel(const float* __restrict__ pos, int Np, int m,
                           int* __restrict__ idx, float* __restrict__ dist)
{
    int b = blockIdx.x;
    const float* P = pos + (size_t)b * Np * 3;
    float* D = dist + (size_t)b * Np;
    int t = threadIdx.x;
    const int T = blockDim.x;
    __shared__ float sv[256];
    __shared__ int   si[256];
    __shared__ int   s_cur;

    float x0 = P[0], y0 = P[1], z0 = P[2];
    for (int j = t; j < Np; j += T) {
        float dx = P[3*j] - x0, dy = P[3*j+1] - y0, dz = P[3*j+2] - z0;
        D[j] = dx*dx + dy*dy + dz*dz;
    }
    if (t == 0) idx[b*m] = 0;
    __syncthreads();

    for (int i = 1; i < m; ++i) {
        float bv = -1.0f; int bi = 0;
        for (int j = t; j < Np; j += T) {
            float v = D[j];
            if (v > bv) { bv = v; bi = j; }
        }
        sv[t] = bv; si[t] = bi;
        __syncthreads();
        for (int s = T/2; s > 0; s >>= 1) {
            if (t < s) {
                if (sv[t+s] > sv[t] || (sv[t+s] == sv[t] && si[t+s] < si[t])) {
                    sv[t] = sv[t+s]; si[t] = si[t+s];
                }
            }
            __syncthreads();
        }
        if (t == 0) { s_cur = si[0]; idx[b*m + i] = si[0]; }
        __syncthreads();
        int j0 = s_cur;
        float cx = P[3*j0], cy = P[3*j0+1], cz = P[3*j0+2];
        for (int j = t; j < Np; j += T) {
            float dx = P[3*j] - cx, dy = P[3*j+1] - cy, dz = P[3*j+2] - cz;
            float d = dx*dx + dy*dy + dz*dz;
            if (d < D[j]) D[j] = d;
        }
        __syncthreads();
    }
}

// ---------------------------------------------------------------------------
// Gather 3D points by index.
// ---------------------------------------------------------------------------
__global__ void gather3(const float* __restrict__ src, const int* __restrict__ idx,
                        int m, int Np, float* __restrict__ out)
{
    int i = blockIdx.x * blockDim.x + threadIdx.x;
    if (i >= B_ * m) return;
    int b = i / m;
    int j = idx[i];
    const float* s = src + ((size_t)b * Np + j) * 3;
    float* o = out + (size_t)i * 3;
    o[0] = s[0]; o[1] = s[1]; o[2] = s[2];
}

// ---------------------------------------------------------------------------
// Ball query: 64 smallest-d2 neighbors per center; mask = (d2 <= r^2).
// ---------------------------------------------------------------------------
__global__ void ball_topk(const float* __restrict__ pos, const float* __restrict__ ctr,
                          int Np, int M, float r2,
                          int* __restrict__ cols, float* __restrict__ mask)
{
    __shared__ unsigned selbits[512];       // up to 16384 points
    __shared__ float sv[256];
    __shared__ int   si[256];
    int t = threadIdx.x;
    const int T = blockDim.x;
    for (int i = t; i < 512; i += T) selbits[i] = 0u;

    int bm = blockIdx.x;
    int b  = bm / M;
    const float* P = pos + (size_t)b * Np * 3;
    const float* C = ctr + (size_t)bm * 3;
    float cx = C[0], cy = C[1], cz = C[2];
    __syncthreads();

    for (int k = 0; k < NB_; ++k) {
        float bv = 3.0e38f; int bi = -1;
        for (int j = t; j < Np; j += T) {
            if ((selbits[j >> 5] >> (j & 31)) & 1u) continue;
            float dx = P[3*j] - cx, dy = P[3*j+1] - cy, dz = P[3*j+2] - cz;
            float d = dx*dx + dy*dy + dz*dz;
            if (d < bv || (d == bv && j < bi)) { bv = d; bi = j; }
        }
        sv[t] = bv; si[t] = bi;
        __syncthreads();
        for (int s = T/2; s > 0; s >>= 1) {
            if (t < s) {
                if (sv[t+s] < sv[t] || (sv[t+s] == sv[t] && si[t+s] < si[t])) {
                    sv[t] = sv[t+s]; si[t] = si[t+s];
                }
            }
            __syncthreads();
        }
        if (t == 0) {
            int j = si[0];
            cols[(size_t)bm * NB_ + k] = j;
            mask[(size_t)bm * NB_ + k] = (sv[0] <= r2) ? 1.0f : 0.0f;
            selbits[j >> 5] |= 1u << (j & 31);
        }
        __syncthreads();
    }
}

// ---------------------------------------------------------------------------
// SA1 features, padded to 32 cols: rel (3) then zeros.
// ---------------------------------------------------------------------------
__global__ void feat_sa1(const float* __restrict__ pos, const float* __restrict__ ctr,
                         const int* __restrict__ cols, int Np, int M, float* __restrict__ out)
{
    size_t tot = (size_t)B_ * M * NB_ * 32;
    size_t i = (size_t)blockIdx.x * blockDim.x + threadIdx.x;
    if (i >= tot) return;
    size_t row = i >> 5;
    int c = (int)(i & 31);
    float v = 0.f;
    if (c < 3) {
        int bm = (int)(row / NB_);
        int b  = bm / M;
        int j  = cols[row];
        v = pos[((size_t)b * Np + j) * 3 + c] - ctr[(size_t)bm * 3 + c];
    }
    out[i] = v;
}

// ---------------------------------------------------------------------------
// SA2 features padded to Cp cols: x_src[col] (C), rel (3), zeros.
// ---------------------------------------------------------------------------
__global__ void feat_sa2(const float* __restrict__ pos, const float* __restrict__ ctr,
                         const int* __restrict__ cols, const float* __restrict__ xsrc,
                         int C, int Cp, int Np, int M, float* __restrict__ out)
{
    size_t tot = (size_t)B_ * M * NB_ * Cp;
    size_t i = (size_t)blockIdx.x * blockDim.x + threadIdx.x;
    if (i >= tot) return;
    size_t row = i / Cp;
    int c = (int)(i % Cp);
    int bm = (int)(row / NB_);
    int b  = bm / M;
    int j  = cols[row];
    float v = 0.f;
    if (c < C)          v = xsrc[((size_t)b * Np + j) * C + c];
    else if (c < C + 3) v = pos[((size_t)b * Np + j) * 3 + (c - C)] - ctr[(size_t)bm * 3 + (c - C)];
    out[i] = v;
}

// ---------------------------------------------------------------------------
// Weight pre-conversion: f32 [K,C] -> f16 [Kp,Cp], zero-padded.
// ---------------------------------------------------------------------------
__global__ void convert_w(const float* __restrict__ W, int K, int C, int Kp, int Cp,
                          _Float16* __restrict__ out)
{
    size_t tot = (size_t)Kp * Cp;
    size_t i = (size_t)blockIdx.x * blockDim.x + threadIdx.x;
    if (i >= tot) return;
    int k = (int)(i / Cp), c = (int)(i % Cp);
    out[i] = (k < K && c < C) ? (_Float16)W[(size_t)k * C + c] : (_Float16)0.f;
}

// ---------------------------------------------------------------------------
// WMMA GEMM, guard-free software-pipelined hot loop.
//   out[M,Cp] = A[M,Kp] @ Wh[Kp,Cp] (+bias/BN/ReLU)
// Preconditions: M%16==0, Kp%32==0, Cp%(16*NW)==0, pads zero-filled.
// One wave computes a 16 x (16*NW) tile. Double-buffered A/B registers:
// each iteration issues the NEXT K-step's loads before running the current
// WMMA chain, so load latency is covered by a full WMMA chain and the A
// converts never WAR-hazard the in-flight WMMA A operand.
// mode: 0 = bias, 1 = bias+BN+ReLU, 2 = bias+ReLU
// ---------------------------------------------------------------------------
template<int NW>
__global__ void wmma_mlp(const float* __restrict__ A, int M, int Kp,
                         const _Float16* __restrict__ Wh, int Cp,
                         const float* __restrict__ bias, int CoutReal,
                         const float* __restrict__ rm, const float* __restrict__ rv,
                         const float* __restrict__ gam, const float* __restrict__ bet,
                         int mode, float* __restrict__ out)
{
    int lane = threadIdx.x;                       // 0..31
    long tile = (long)blockIdx.x * blockDim.y + threadIdx.y;
    int  tn = Cp / (16 * NW);
    long tm = (long)(M >> 4);
    if (tile >= tm * tn) return;
    int n0 = (int)(tile % tn) * (16 * NW);
    int m0 = (int)(tile / tn) << 4;

    int lh = lane >> 4;        // K-half selector (A), row offset (C/D)
    int lr = lane & 15;

    v8f acc[NW];
    #pragma unroll
    for (int w = 0; w < NW; ++w) acc[w] = (v8f){0.f,0.f,0.f,0.f,0.f,0.f,0.f,0.f};

    // Per-lane bases: A row m0+lr, K-halves offset by 8*lh; B column n0+lr, K-halves offset by 8*lh.
    const float*    Arow = A  + (size_t)(m0 + lr) * Kp + lh * 8;
    const _Float16* Wl   = Wh + (size_t)(lh * 8) * Cp + n0 + lr;

    // ---- pipeline prologue: step 0 loads ----
    float4 a0 = *(const float4*)(Arow);
    float4 a1 = *(const float4*)(Arow + 4);
    float4 a2 = *(const float4*)(Arow + 16);
    float4 a3 = *(const float4*)(Arow + 20);
    v16h bf[NW];
    #pragma unroll
    for (int w = 0; w < NW; ++w) for (int s = 0; s < 16; ++s) bf[w][s] = Wl[(size_t)((s & 7) + ((s >> 3) << 4)) * Cp + 16 * w];

    #pragma unroll 2
    for (int kk = 32; kk < Kp; kk += 32) {
        // issue next step's loads first
        float4 c0 = *(const float4*)(Arow + kk);
        float4 c1 = *(const float4*)(Arow + kk + 4);
        float4 c2 = *(const float4*)(Arow + kk + 16);
        float4 c3 = *(const float4*)(Arow + kk + 20);
        const _Float16* Wn = Wl + (size_t)kk * Cp;
        v16h bn[NW];
        #pragma unroll
        for (int w = 0; w < NW; ++w) for (int s = 0; s < 16; ++s) bn[w][s] = Wn[(size_t)((s & 7) + ((s >> 3) << 4)) * Cp + 16 * w];
        __builtin_prefetch(Arow + kk + 32, 0, 3);

        // compute current step
        v16h af = cvt_a16(a0, a1, a2, a3);
        #pragma unroll
        for (int w = 0; w < NW; ++w) {
            acc[w] = __builtin_amdgcn_wmma_f32_16x16x32_f16(false, af, false, bf[w],
                                                            (short)0, acc[w], false, false);
        }
        // rotate double buffers (resolved by unrolling into register renaming)
        a0 = c0; a1 = c1; a2 = c2; a3 = c3;
        #pragma unroll
        for (int w = 0; w < NW; ++w) bf[w] = bn[w];
    }

    // ---- pipeline epilogue: final K-step ----
    {
        v16h af = cvt_a16(a0, a1, a2, a3);
        #pragma unroll
        for (int w = 0; w < NW; ++w) {
            acc[w] = __builtin_amdgcn_wmma_f32_16x16x32_f16(false, af, false, bf[w],
                                                            (short)0, acc[w], false, false);
        }
    }

    // Epilogue: lane -> column n + 16w; VGPR r -> row m0 + r + 8*lh.
    #pragma unroll
    for (int w = 0; w < NW; ++w) {
        int n  = n0 + lr + 16 * w;
        int nb = (n < CoutReal) ? n : 0;
        float bia = (n < CoutReal) ? bias[n] : 0.f;
        float sc = 1.f, sh = 0.f;
        if (mode == 1) {
            float a = rsqrtf(rv[nb] + 1e-5f) * gam[nb];
            sc = a; sh = bet[nb] - rm[nb] * a;
        }
        #pragma unroll
        for (int r = 0; r < 8; ++r) {
            float v = acc[w][r] + bia;
            if (mode == 1) v = v * sc + sh;
            if (mode >= 1) v = fmaxf(v, 0.f);
            out[(size_t)(m0 + r + lh * 8) * Cp + n] = v;
        }
    }
}

// ---------------------------------------------------------------------------
// Masked max-pool over NB_ neighbors: h[B*M*NB, C] -> out[B*M, C]
// ---------------------------------------------------------------------------
__global__ void masked_maxpool(const float* __restrict__ h, const float* __restrict__ mask,
                               int M, int C, float* __restrict__ out)
{
    size_t tot = (size_t)B_ * M * C;
    size_t i = (size_t)blockIdx.x * blockDim.x + threadIdx.x;
    if (i >= tot) return;
    size_t bm = i / C;
    int c = (int)(i % C);
    const float* hp = h + (bm * NB_) * (size_t)C + c;
    float best = -3.4e38f;
    for (int k = 0; k < NB_; ++k) {
        if (mask[bm * NB_ + k] > 0.5f) {
            float v = hp[(size_t)k * C];
            if (v > best) best = v;
        }
    }
    out[i] = best;
}

// ---------------------------------------------------------------------------
// Row-wise concat of two feature maps into a (possibly padded) Cp-wide buffer.
// ---------------------------------------------------------------------------
__global__ void concat2(const float* __restrict__ a, int Ca,
                        const float* __restrict__ b, int Cb,
                        int Cp, size_t R, float* __restrict__ out)
{
    size_t tot = R * (size_t)Cp;
    size_t i = (size_t)blockIdx.x * blockDim.x + threadIdx.x;
    if (i >= tot) return;
    size_t r = i / Cp;
    int c = (int)(i % Cp);
    float v = 0.f;
    if (c < Ca)           v = a[r * Ca + c];
    else if (c < Ca + Cb) v = b[r * Cb + (c - Ca)];
    out[i] = v;
}

// ---------------------------------------------------------------------------
// Global max over R points per batch: x[B,R,C] -> out[B,C]
// ---------------------------------------------------------------------------
__global__ void rowmax(const float* __restrict__ x, int R, int C, float* __restrict__ out)
{
    int i = blockIdx.x * blockDim.x + threadIdx.x;
    if (i >= B_ * C) return;
    int b = i / C, c = i % C;
    const float* p = x + ((size_t)b * R) * C + c;
    float best = -3.4e38f;
    for (int r = 0; r < R; ++r) {
        float v = p[(size_t)r * C];
        if (v > best) best = v;
    }
    out[i] = best;
}

// ---------------------------------------------------------------------------
// Broadcast global vector + concat per-point features (fp3 input).
// ---------------------------------------------------------------------------
__global__ void bcast_concat(const float* __restrict__ g, int Cg,
                             const float* __restrict__ x, int Cx,
                             int M, float* __restrict__ out)
{
    int C = Cg + Cx;
    size_t tot = (size_t)B_ * M * C;
    size_t i = (size_t)blockIdx.x * blockDim.x + threadIdx.x;
    if (i >= tot) return;
    size_t bm = i / C;
    int c = (int)(i % C);
    int b = (int)(bm / M);
    out[i] = (c < Cg) ? g[(size_t)b * Cg + c] : x[bm * Cx + (c - Cg)];
}

// ---------------------------------------------------------------------------
// 3-NN indices + inverse-d2 weights from src cloud to dst cloud.
// ---------------------------------------------------------------------------
__global__ void knn3(const float* __restrict__ srcp, int Ms,
                     const float* __restrict__ dstp, int Md,
                     int* __restrict__ idx, float* __restrict__ w)
{
    int i = blockIdx.x * blockDim.x + threadIdx.x;
    if (i >= B_ * Md) return;
    int b = i / Md;
    const float* S = srcp + (size_t)b * Ms * 3;
    const float* D = dstp + (size_t)i * 3;
    float px = D[0], py = D[1], pz = D[2];
    float d0 = 3.4e38f, d1 = 3.4e38f, d2 = 3.4e38f;
    int   i0 = 0, i1 = 0, i2 = 0;
    for (int j = 0; j < Ms; ++j) {
        float dx = S[3*j] - px, dy = S[3*j+1] - py, dz = S[3*j+2] - pz;
        float d = dx*dx + dy*dy + dz*dz;
        if      (d < d0) { d2 = d1; i2 = i1; d1 = d0; i1 = i0; d0 = d; i0 = j; }
        else if (d < d1) { d2 = d1; i2 = i1; d1 = d;  i1 = j; }
        else if (d < d2) { d2 = d;  i2 = j; }
    }
    idx[3*i]   = i0; idx[3*i+1] = i1; idx[3*i+2] = i2;
    w[3*i]   = 1.f / fmaxf(d0, 1e-16f);
    w[3*i+1] = 1.f / fmaxf(d1, 1e-16f);
    w[3*i+2] = 1.f / fmaxf(d2, 1e-16f);
}

// ---------------------------------------------------------------------------
// Apply 3-NN interpolation: out[b,j,c] = sum_k w*f[idx] / sum_k w
// ---------------------------------------------------------------------------
__global__ void knn_apply(const float* __restrict__ f, int Ms, int C,
                          const int* __restrict__ idx, const float* __restrict__ w,
                          int Md, float* __restrict__ out)
{
    size_t tot = (size_t)B_ * Md * C;
    size_t i = (size_t)blockIdx.x * blockDim.x + threadIdx.x;
    if (i >= tot) return;
    size_t bd = i / C;
    int c = (int)(i % C);
    int b = (int)(bd / Md);
    const float* F = f + (size_t)b * Ms * C;
    float ws = 0.f, acc = 0.f;
    #pragma unroll
    for (int k = 0; k < 3; ++k) {
        float wk = w[bd * 3 + k];
        acc += wk * F[(size_t)idx[bd * 3 + k] * C + c];
        ws  += wk;
    }
    out[i] = acc / ws;
}

// ---------------------------------------------------------------------------
// log_softmax over the 13 class logits per row (input stride NCP_ padded).
// ---------------------------------------------------------------------------
__global__ void log_softmax13(const float* __restrict__ x, float* __restrict__ out)
{
    size_t i = (size_t)blockIdx.x * blockDim.x + threadIdx.x;
    if (i >= (size_t)B_ * N_) return;
    const float* p = x + i * NCP_;
    float mx = p[0];
    #pragma unroll
    for (int c = 1; c < NC_; ++c) mx = fmaxf(mx, p[c]);
    float s = 0.f;
    #pragma unroll
    for (int c = 0; c < NC_; ++c) s += expf(p[c] - mx);
    float l = logf(s) + mx;
    float* o = out + i * NC_;
    #pragma unroll
    for (int c = 0; c < NC_; ++c) o[c] = p[c] - l;
}

// ===========================================================================
// Host orchestration
// ===========================================================================
struct Layer {
    const float *W, *b, *rm, *rv, *g, *bt;
    _Float16* Wh;
    int K, C, Kp, Cp;
};

extern "C" void kernel_launch(void* const* d_in, const int* in_sizes, int n_in,
                              void* d_out, int out_size, void* d_ws, size_t ws_size,
                              hipStream_t stream)
{
    (void)in_sizes; (void)n_in; (void)out_size; (void)ws_size;

    const float* pos = (const float*)d_in[0];

    // ---- workspace bump allocator ----
    char* wp = (char*)d_ws;
    size_t off = 0;
    auto alloc = [&](size_t bytes) -> void* {
        void* p = wp + off;
        off += (bytes + 255) & ~(size_t)255;
        return p;
    };

    // ---- parameter unpacking (setup_inputs insertion order) + f16 prep ----
    int ip = 1;
    auto take = [&](bool bn, int K, int C) -> Layer {
        Layer L;
        L.W = (const float*)d_in[ip++];
        L.b = (const float*)d_in[ip++];
        if (bn) {
            L.rm = (const float*)d_in[ip++];
            L.rv = (const float*)d_in[ip++];
            L.g  = (const float*)d_in[ip++];
            L.bt = (const float*)d_in[ip++];
        } else { L.rm = L.rv = L.g = L.bt = nullptr; }
        L.K = K; L.C = C;
        L.Kp = (K + 31) & ~31;
        L.Cp = (C + 15) & ~15;
        L.Wh = (_Float16*)alloc((size_t)L.Kp * L.Cp * 2);
        convert_w<<<gridFor((size_t)L.Kp * L.Cp, 256), 256, 0, stream>>>(
            L.W, K, C, L.Kp, L.Cp, L.Wh);
        return L;
    };
    Layer sa1[3]  = { take(true,3,64),     take(true,64,64),   take(false,64,128)  };
    Layer sa2[3]  = { take(true,131,128),  take(true,128,128), take(false,128,256) };
    Layer sa3[3]  = { take(true,259,256),  take(true,256,512), take(false,512,1024)};
    Layer sfp3[2] = { take(true,1280,256), take(false,256,256) };
    Layer sfp2[2] = { take(true,384,256),  take(false,256,128) };
    Layer sfp1[3] = { take(true,128,128),  take(true,128,128), take(false,128,128) };
    Layer ifp3[2] = { take(true,1280,256), take(false,256,256) };
    Layer ifp2[2] = { take(true,384,256),  take(false,256,128) };
    Layer ifp1[3] = { take(true,128,128),  take(true,128,128), take(false,128,64)  };
    Layer cls[3]  = { take(false,128,128), take(false,128,128), take(false,128,NC_) };

    float* fdist = (float*)alloc((size_t)B_*N_*4);
    int*   idx1  = (int*)  alloc((size_t)B_*M1_*4);
    float* p1    = (float*)alloc((size_t)B_*M1_*3*4);
    int*   cols1 = (int*)  alloc((size_t)B_*M1_*NB_*4);
    float* msk1  = (float*)alloc((size_t)B_*M1_*NB_*4);
    float* feat1 = (float*)alloc((size_t)B_*M1_*NB_*32*4);    // Kp=32 padded
    float* x1    = (float*)alloc((size_t)B_*M1_*128*4);
    int*   idx2  = (int*)  alloc((size_t)B_*M2_*4);
    float* p2    = (float*)alloc((size_t)B_*M2_*3*4);
    int*   cols2 = (int*)  alloc((size_t)B_*M2_*NB_*4);
    float* msk2  = (float*)alloc((size_t)B_*M2_*NB_*4);
    float* feat2 = (float*)alloc((size_t)B_*M2_*NB_*160*4);   // Kp=160 padded
    float* x2    = (float*)alloc((size_t)B_*M2_*256*4);
    float* cat3  = (float*)alloc((size_t)B_*M2_*288*4);       // Kp=288 padded
    float* gbuf  = (float*)alloc((size_t)B_*1024*4);
    float* cat13 = (float*)alloc((size_t)B_*M2_*1280*4);
    float* f3    = (float*)alloc((size_t)B_*M2_*256*4);
    float* y2    = (float*)alloc((size_t)B_*M1_*256*4);
    float* catf2 = (float*)alloc((size_t)B_*M1_*384*4);
    float* f2    = (float*)alloc((size_t)B_*M1_*128*4);
    int*   ki2   = (int*)  alloc((size_t)B_*M1_*3*4);
    float* kw2   = (float*)alloc((size_t)B_*M1_*3*4);
    int*   ki1   = (int*)  alloc((size_t)B_*N_*3*4);
    float* kw1   = (float*)alloc((size_t)B_*N_*3*4);
    float* logit = (float*)alloc((size_t)B_*N_*NCP_*4);
    float* G0    = (float*)alloc((size_t)B_*M1_*NB_*128*4);   // 86 MB ping
    float* G1    = (float*)alloc((size_t)B_*M1_*NB_*128*4);   // 86 MB pong

    auto mlp_layer = [&](const float* A, long M, const Layer& L, int mode, float* out) {
        if (L.Cp % 64 == 0) {
            long tiles = (M >> 4) * (long)(L.Cp / 64);
            wmma_mlp<4><<<(unsigned)((tiles + 7) / 8), dim3(32, 8), 0, stream>>>(
                A, (int)M, L.Kp, L.Wh, L.Cp, L.b, L.C, L.rm, L.rv, L.g, L.bt, mode, out);
        } else {
            long tiles = (M >> 4) * (long)(L.Cp / 16);
            wmma_mlp<1><<<(unsigned)((tiles + 7) / 8), dim3(32, 8), 0, stream>>>(
                A, (int)M, L.Kp, L.Wh, L.Cp, L.b, L.C, L.rm, L.rv, L.g, L.bt, mode, out);
        }
    };

    // ================= SA1 =================
    fps_kernel<<<B_, 256, 0, stream>>>(pos, N_, M1_, idx1, fdist);
    gather3<<<gridFor((size_t)B_*M1_, 256), 256, 0, stream>>>(pos, idx1, M1_, N_, p1);
    ball_topk<<<B_*M1_, 256, 0, stream>>>(pos, p1, N_, M1_, 0.04f, cols1, msk1);
    feat_sa1<<<gridFor((size_t)B_*M1_*NB_*32, 256), 256, 0, stream>>>(pos, p1, cols1, N_, M1_, feat1);
    long R1 = (long)B_ * M1_ * NB_;
    mlp_layer(feat1, R1, sa1[0], 1, G0);
    mlp_layer(G0,    R1, sa1[1], 1, G1);
    mlp_layer(G1,    R1, sa1[2], 0, G0);
    masked_maxpool<<<gridFor((size_t)B_*M1_*128, 256), 256, 0, stream>>>(G0, msk1, M1_, 128, x1);

    // ================= SA2 =================
    fps_kernel<<<B_, 256, 0, stream>>>(p1, M1_, M2_, idx2, fdist);
    gather3<<<gridFor((size_t)B_*M2_, 256), 256, 0, stream>>>(p1, idx2, M2_, M1_, p2);
    ball_topk<<<B_*M2_, 256, 0, stream>>>(p1, p2, M1_, M2_, 0.16f, cols2, msk2);
    feat_sa2<<<gridFor((size_t)B_*M2_*NB_*160, 256), 256, 0, stream>>>(p1, p2, cols2, x1, 128, 160, M1_, M2_, feat2);
    long R2 = (long)B_ * M2_ * NB_;
    mlp_layer(feat2, R2, sa2[0], 1, G0);
    mlp_layer(G0,    R2, sa2[1], 1, G1);
    mlp_layer(G1,    R2, sa2[2], 0, G0);
    masked_maxpool<<<gridFor((size_t)B_*M2_*256, 256), 256, 0, stream>>>(G0, msk2, M2_, 256, x2);

    // ================= SA3 (global) =================
    concat2<<<gridFor((size_t)B_*M2_*288, 256), 256, 0, stream>>>(x2, 256, p2, 3, 288, (size_t)B_*M2_, cat3);
    long R3 = (long)B_ * M2_;
    mlp_layer(cat3, R3, sa3[0], 1, G0);
    mlp_layer(G0,   R3, sa3[1], 1, G1);
    mlp_layer(G1,   R3, sa3[2], 0, G0);
    rowmax<<<gridFor((size_t)B_*1024, 256), 256, 0, stream>>>(G0, M2_, 1024, gbuf);

    // ================= kNN graphs (shared by both branches) =================
    knn3<<<gridFor((size_t)B_*M1_, 256), 256, 0, stream>>>(p2, M2_, p1, M1_, ki2, kw2);
    knn3<<<gridFor((size_t)B_*N_,  256), 256, 0, stream>>>(p1, M1_, pos, N_, ki1, kw1);

    // fp3 input (same for both branches): [g | x2] -> [B,M2,1280]
    bcast_concat<<<gridFor((size_t)B_*M2_*1280, 256), 256, 0, stream>>>(gbuf, 1024, x2, 256, M2_, cat13);

    long R4 = (long)B_ * M1_;
    long R5 = (long)B_ * N_;
    for (int br = 0; br < 2; ++br) {
        const Layer* fp3 = br ? ifp3 : sfp3;
        const Layer* fp2 = br ? ifp2 : sfp2;
        const Layer* fp1 = br ? ifp1 : sfp1;

        // FP3: 1280 -> 256 (BN+ReLU) -> 256
        mlp_layer(cat13, R3, fp3[0], 1, G0);
        mlp_layer(G0,    R3, fp3[1], 0, f3);
        // FP2: interp(p2 -> p1), concat x1, 384 -> 256 (BN+ReLU) -> 128
        knn_apply<<<gridFor((size_t)B_*M1_*256, 256), 256, 0, stream>>>(f3, M2_, 256, ki2, kw2, M1_, y2);
        concat2<<<gridFor((size_t)B_*M1_*384, 256), 256, 0, stream>>>(y2, 256, x1, 128, 384, (size_t)B_*M1_, catf2);
        mlp_layer(catf2, R4, fp2[0], 1, G0);
        mlp_layer(G0,    R4, fp2[1], 0, f2);
        // FP1: interp(p1 -> pos), 128 -> 128 -> 128 -> {128 | 64}
        knn_apply<<<gridFor((size_t)B_*N_*128, 256), 256, 0, stream>>>(f2, M1_, 128, ki1, kw1, N_, G0);
        mlp_layer(G0, R5, fp1[0], 1, G1);
        mlp_layer(G1, R5, fp1[1], 1, G0);
        if (br == 0) {
            mlp_layer(G0, R5, fp1[2], 0, G1);      // sem_x
            mlp_layer(G1, R5, cls[0], 2, G0);
            mlp_layer(G0, R5, cls[1], 2, G1);
            mlp_layer(G1, R5, cls[2], 0, logit);   // 13 -> padded 16
            log_softmax13<<<gridFor((size_t)B_*N_, 256), 256, 0, stream>>>(logit, (float*)d_out);
        } else {
            mlp_layer(G0, R5, fp1[2], 0, (float*)d_out + (size_t)B_*N_*NC_);
        }
    }
}
